// Decoder_53712861004005
// MI455X (gfx1250) — matrix operand order repaired and checked by reading, not compiled, not run on hardware
//
#include <hip/hip_runtime.h>
#include <math.h>

// Problem dimensions (from reference setup_inputs)
#define BATCH 64
#define TSTEPS 64
#define VOCAB 32000
#define DDEC 1024
#define DEMB 512

typedef __attribute__((ext_vector_type(8)))  __bf16 v8bf;
typedef __attribute__((ext_vector_type(16))) __bf16 v16bf;
typedef __attribute__((ext_vector_type(8)))  float  v8f;

// ---------------------------------------------------------------------------
// WMMA fragment loaders for V_WMMA_F32_16X16X32_BF16 (wave32).
// A (16x32, MxK), row-major source [*, lda]:
//   lanes 0-15: M=lane, VGPR0-3 hold K = k0+half*8 + 0..7 (pairs),
//               VGPR4-7 hold K = k0+16+half*8 + 0..7, half = lane>>4.
// B (32x16, KxN) built from row-major weight W[N,K] (i.e. B[k][n] = W[nb+n][k]):
//   lane n (0-15) / n+16: VGPR0-3 hold K = k0+half*8 + 0..7,
//   VGPR4-7 hold K = k0+16+half*8 + 0..7 (same order as A).
// ---------------------------------------------------------------------------
static __device__ __forceinline__ v16bf load_a16(const __bf16* __restrict__ A,
                                                 int lda, int mbase, int k0, int lane) {
    const int row = mbase + (lane & 15);
    const int h8  = (lane >> 4) << 3;   // 0 or 8
    const __bf16* p = A + (size_t)row * lda + k0 + h8;
    v8bf lo = *(const v8bf*)p;
    v8bf hi = *(const v8bf*)(p + 16);
    v16bf r;
#pragma unroll
    for (int i = 0; i < 8; ++i) { r[i] = lo[i]; r[i + 8] = hi[i]; }
    return r;
}

static __device__ __forceinline__ v16bf load_b16(const __bf16* __restrict__ W,
                                                 int ldw, int nbase, int k0, int lane) {
    const int n   = lane & 15;
    const int h16 = (lane >> 4) << 3;   // 0 or 8
    const __bf16* p = W + (size_t)(nbase + n) * ldw + k0 + h16;
    v8bf lo = *(const v8bf*)p;
    v8bf hi = *(const v8bf*)(p + 16);
    v16bf r;
#pragma unroll
    for (int i = 0; i < 8; ++i) { r[i] = lo[i]; r[i + 8] = hi[i]; }
    return r;
}

#define WMMA_BF16(A, Bv, C) \
    __builtin_amdgcn_wmma_f32_16x16x32_bf16(false, (A), false, (Bv), (short)0, (C), false, false)

// ---------------------------------------------------------------------------
// fp32 -> bf16 weight conversion (done once; weights then live in L2 in bf16)
// ---------------------------------------------------------------------------
__global__ void f2bf_kernel(const float* __restrict__ src, __bf16* __restrict__ dst, size_t n) {
    size_t i = (size_t)blockIdx.x * blockDim.x + threadIdx.x;
    size_t stride = (size_t)gridDim.x * blockDim.x;
    for (; i < n; i += stride) dst[i] = (__bf16)src[i];
}

// h0 = broadcast(init); x = embs[bos]  (bf16 copy of h kept for WMMA A-operand)
__global__ void init_kernel(const float* __restrict__ init,
                            const float* __restrict__ embs,
                            const int* __restrict__ bos_idx,
                            float* __restrict__ h_f, __bf16* __restrict__ h_b,
                            __bf16* __restrict__ xb) {
    const int bos = bos_idx[0];
    const int i = blockIdx.x * blockDim.x + threadIdx.x;
    const int stride = gridDim.x * blockDim.x;
    for (int j = i; j < BATCH * DDEC; j += stride) {
        float v = init[j % DDEC];
        h_f[j] = v;
        h_b[j] = (__bf16)v;
    }
    for (int j = i; j < BATCH * DEMB; j += stride)
        xb[j] = (__bf16)embs[(size_t)bos * DEMB + (j % DEMB)];
}

// ---------------------------------------------------------------------------
// Fused GRU cell: one wave per 16x16 tile of h (4 m-tiles x 64 n-tiles = 256
// waves). Each wave holds 6 fp32 accumulators (r/z/n gates for both GEMMs)
// sharing each A fragment, then applies the gate nonlinearity in-register.
// ---------------------------------------------------------------------------
__global__ __launch_bounds__(256) void gru_kernel(
    const __bf16* __restrict__ xb,   // [BATCH, DEMB]
    const __bf16* __restrict__ hbp,  // [BATCH, DDEC] bf16 prev state
    const float*  __restrict__ hp,   // [BATCH, DDEC] fp32 prev state
    const __bf16* __restrict__ wih,  // [3*DDEC, DEMB]
    const __bf16* __restrict__ whh,  // [3*DDEC, DDEC]
    const float*  __restrict__ bih,  // [3*DDEC]
    const float*  __restrict__ bhh,  // [3*DDEC]
    float*  __restrict__ hn_f,       // [BATCH, DDEC]
    __bf16* __restrict__ hn_b) {
    const int lane  = threadIdx.x & 31;
    const int wid   = (blockIdx.x * blockDim.x + threadIdx.x) >> 5;
    const int mbase = (wid >> 6) << 4;   // 0..3 -> row tile
    const int nbase = (wid & 63) << 4;   // 0..63 -> d_dec column tile

    v8f air = {}, aiz = {}, ain = {}, ahr = {}, ahz = {}, ahn = {};

    // gi = x @ w_ih^T   (K = 512)
    for (int k0 = 0; k0 < DEMB; k0 += 32) {
        v16bf a  = load_a16(xb, DEMB, mbase, k0, lane);
        v16bf br = load_b16(wih, DEMB, nbase,            k0, lane);
        v16bf bz = load_b16(wih, DEMB, DDEC + nbase,     k0, lane);
        v16bf bn = load_b16(wih, DEMB, 2 * DDEC + nbase, k0, lane);
        air = WMMA_BF16(a, br, air);
        aiz = WMMA_BF16(a, bz, aiz);
        ain = WMMA_BF16(a, bn, ain);
    }
    // gh = h @ w_hh^T   (K = 1024)
    for (int k0 = 0; k0 < DDEC; k0 += 32) {
        v16bf a  = load_a16(hbp, DDEC, mbase, k0, lane);
        v16bf br = load_b16(whh, DDEC, nbase,            k0, lane);
        v16bf bz = load_b16(whh, DDEC, DDEC + nbase,     k0, lane);
        v16bf bn = load_b16(whh, DDEC, 2 * DDEC + nbase, k0, lane);
        ahr = WMMA_BF16(a, br, ahr);
        ahz = WMMA_BF16(a, bz, ahz);
        ahn = WMMA_BF16(a, bn, ahn);
    }

    // C/D layout: VGPR i -> row mbase + half*8 + i, col nbase + (lane&15)
    const int col   = nbase + (lane & 15);
    const int rbase = mbase + ((lane >> 4) << 3);
    const float bir = bih[col], biz = bih[DDEC + col], bin = bih[2 * DDEC + col];
    const float bhr = bhh[col], bhz = bhh[DDEC + col], bhn = bhh[2 * DDEC + col];
#pragma unroll
    for (int i = 0; i < 8; ++i) {
        const int row = rbase + i;
        const float r = 1.0f / (1.0f + expf(-(air[i] + bir + ahr[i] + bhr)));
        const float z = 1.0f / (1.0f + expf(-(aiz[i] + biz + ahz[i] + bhz)));
        const float n = tanhf(ain[i] + bin + r * (ahn[i] + bhn));
        const float ho = hp[(size_t)row * DDEC + col];
        const float hv = (1.0f - z) * n + z * ho;
        hn_f[(size_t)row * DDEC + col] = hv;
        hn_b[(size_t)row * DDEC + col] = (__bf16)hv;
    }
}

// ---------------------------------------------------------------------------
// logits[b, t, :] = h @ W_out^T + b_out.
// One wave per 16-column vocab tile (2000 waves); each wave carries all four
// 16-row batch tiles so every W_out fragment is loaded exactly once (4x less
// B traffic from L2 than one-tile-per-wave). A fragments (128 KB bf16 state)
// are shared by all waves on a WGP and hit WGP$.
// ---------------------------------------------------------------------------
__global__ __launch_bounds__(256) void logits_kernel(
    const __bf16* __restrict__ hb,    // [BATCH, DDEC]
    const __bf16* __restrict__ wout,  // [VOCAB, DDEC]
    const float*  __restrict__ bout,  // [VOCAB]
    float* __restrict__ logits,       // [BATCH, TSTEPS, VOCAB]
    int t) {
    const int lane  = threadIdx.x & 31;
    const int wid   = (blockIdx.x * blockDim.x + threadIdx.x) >> 5;  // 0..1999
    const int nbase = wid << 4;

    v8f acc0 = {}, acc1 = {}, acc2 = {}, acc3 = {};
    for (int k0 = 0; k0 < DDEC; k0 += 32) {
        v16bf b  = load_b16(wout, DDEC, nbase, k0, lane);
        v16bf a0 = load_a16(hb, DDEC, 0,  k0, lane);
        v16bf a1 = load_a16(hb, DDEC, 16, k0, lane);
        v16bf a2 = load_a16(hb, DDEC, 32, k0, lane);
        v16bf a3 = load_a16(hb, DDEC, 48, k0, lane);
        acc0 = WMMA_BF16(a0, b, acc0);
        acc1 = WMMA_BF16(a1, b, acc1);
        acc2 = WMMA_BF16(a2, b, acc2);
        acc3 = WMMA_BF16(a3, b, acc3);
    }

    const int col   = nbase + (lane & 15);
    const int rsub  = (lane >> 4) << 3;            // 0 or 8 within a 16-row tile
    const float bo  = bout[col];
    float* outc = logits + (size_t)t * VOCAB + col;
#pragma unroll
    for (int i = 0; i < 8; ++i) {
        const int r = rsub + i;
        outc[(size_t)(r)      * TSTEPS * VOCAB] = acc0[i] + bo;
        outc[(size_t)(r + 16) * TSTEPS * VOCAB] = acc1[i] + bo;
        outc[(size_t)(r + 32) * TSTEPS * VOCAB] = acc2[i] + bo;
        outc[(size_t)(r + 48) * TSTEPS * VOCAB] = acc3[i] + bo;
    }
}

// ---------------------------------------------------------------------------
// argmax (first-occurrence, matching jnp.argmax) + embedding gather -> bf16 x
// ---------------------------------------------------------------------------
__global__ __launch_bounds__(256) void argmax_embed_kernel(
    const float* __restrict__ logits, const float* __restrict__ embs,
    __bf16* __restrict__ xb, float* __restrict__ preds, int t) {
    __shared__ float sv[256];
    __shared__ int   si[256];
    const int b   = blockIdx.x;
    const int tid = threadIdx.x;
    const float* row = logits + (size_t)b * TSTEPS * VOCAB + (size_t)t * VOCAB;

    float best = -3.402823466e+38f;
    int bi = 0x7fffffff;
    for (int v = tid; v < VOCAB; v += 256) {
        const float x = row[v];
        if (x > best) { best = x; bi = v; }   // ascending v -> keeps first max
    }
    sv[tid] = best; si[tid] = bi;
    __syncthreads();
    for (int s = 128; s > 0; s >>= 1) {
        if (tid < s) {
            const float ov = sv[tid + s]; const int oi = si[tid + s];
            if (ov > sv[tid] || (ov == sv[tid] && oi < si[tid])) { sv[tid] = ov; si[tid] = oi; }
        }
        __syncthreads();
    }
    const int p = si[0];
    if (tid == 0) preds[(size_t)b * TSTEPS + t] = (float)p;
    for (int e = tid; e < DEMB; e += 256)
        xb[(size_t)b * DEMB + e] = (__bf16)embs[(size_t)p * DEMB + e];
}

// ---------------------------------------------------------------------------
extern "C" void kernel_launch(void* const* d_in, const int* in_sizes, int n_in,
                              void* d_out, int out_size, void* d_ws, size_t ws_size,
                              hipStream_t stream) {
    (void)in_sizes; (void)n_in; (void)out_size; (void)ws_size;

    const float* embs  = (const float*)d_in[0];
    const float* init  = (const float*)d_in[1];
    const float* w_ih  = (const float*)d_in[2];
    const float* w_hh  = (const float*)d_in[3];
    const float* b_ih  = (const float*)d_in[4];
    const float* b_hh  = (const float*)d_in[5];
    const float* W_out = (const float*)d_in[6];
    const float* b_out = (const float*)d_in[7];
    const int*   bos   = (const int*)d_in[8];

    float* logits = (float*)d_out;                               // [B, T, V]
    float* preds  = logits + (size_t)BATCH * TSTEPS * VOCAB;     // [B, T]

    // Workspace carve (256B-aligned): bf16 weights + double-buffered state
    char* ws = (char*)d_ws;
    auto carve = [&](size_t bytes) -> char* {
        char* p = ws; ws += (bytes + 255) & ~(size_t)255; return p;
    };
    __bf16* wih_b  = (__bf16*)carve((size_t)3 * DDEC * DEMB * sizeof(__bf16));
    __bf16* whh_b  = (__bf16*)carve((size_t)3 * DDEC * DDEC * sizeof(__bf16));
    __bf16* wout_b = (__bf16*)carve((size_t)VOCAB * DDEC * sizeof(__bf16));
    float*  h0     = (float*) carve((size_t)BATCH * DDEC * sizeof(float));
    float*  h1     = (float*) carve((size_t)BATCH * DDEC * sizeof(float));
    __bf16* hb0    = (__bf16*)carve((size_t)BATCH * DDEC * sizeof(__bf16));
    __bf16* hb1    = (__bf16*)carve((size_t)BATCH * DDEC * sizeof(__bf16));
    __bf16* xb     = (__bf16*)carve((size_t)BATCH * DEMB * sizeof(__bf16));

    // One-time weight down-conversion (bf16 weights then stay L2-resident)
    f2bf_kernel<<<512,  256, 0, stream>>>(w_ih,  wih_b,  (size_t)3 * DDEC * DEMB);
    f2bf_kernel<<<1024, 256, 0, stream>>>(w_hh,  whh_b,  (size_t)3 * DDEC * DDEC);
    f2bf_kernel<<<4096, 256, 0, stream>>>(W_out, wout_b, (size_t)VOCAB * DDEC);
    init_kernel<<<256,  256, 0, stream>>>(init, embs, bos, h0, hb0, xb);

    for (int t = 0; t < TSTEPS; ++t) {
        const float*  hp  = (t & 1) ? h1  : h0;
        float*        hn  = (t & 1) ? h0  : h1;
        const __bf16* hbp = (t & 1) ? hb1 : hb0;
        __bf16*       hbn = (t & 1) ? hb0 : hb1;

        // 256 waves: 4 m-tiles x 64 n-tiles, 6 fused gate accumulators each
        gru_kernel<<<32, 256, 0, stream>>>(xb, hbp, hp, wih_b, whh_b, b_ih, b_hh, hn, hbn);
        // 2000 waves: one vocab tile each, all 4 batch tiles per wave
        logits_kernel<<<250, 256, 0, stream>>>(hbn, wout_b, b_out, logits, t);
        // per-batch-row argmax + next-token embedding gather
        argmax_embed_kernel<<<BATCH, 256, 0, stream>>>(logits, embs, xb, preds, t);
    }
}
